// LyraGemma3Attention_83047487636025
// MI455X (gfx1250) — matrix operand order repaired and checked by reading, not compiled, not run on hardware
//
#include <hip/hip_runtime.h>

// ---------------- problem constants (match reference) ----------------
constexpr int kS     = 2048;          // sequence length
constexpr int kH     = 16;            // q heads
constexpr int kKV    = 4;             // kv heads
constexpr int kD     = 128;           // head dim
constexpr int kHID   = kH * kD;       // 2048
constexpr int kKVD   = kKV * kD;      // 512
constexpr int kCACHE = 2048;
constexpr int kLW    = kCACHE + kS;   // 4096 lyra keys
constexpr float kEPS   = 1e-6f;
constexpr float kSCALE = 0.08838834764831845f;  // 128^-0.5

// ---------------- CDNA5 vector types ----------------
typedef __attribute__((ext_vector_type(16))) __bf16        v16bf;
typedef __attribute__((ext_vector_type(8)))  unsigned int  v8u;
typedef __attribute__((ext_vector_type(4)))  unsigned int  v4u;
typedef __attribute__((ext_vector_type(8)))  float         v8f;

__device__ __forceinline__ unsigned f2bf(float f) {
    // round-to-nearest-even fp32 -> bf16 bits
    unsigned u = __float_as_uint(f);
    return (u + 0x7FFFu + ((u >> 16) & 1u)) >> 16;
}

// =====================================================================
// bf16 GEMM:  C[M,N] = alpha * A[M,K] @ B[N,K]^T  (+ mask epilogues)
// A, B are bf16; C is fp32 or bf16.
// REQUIRES: M%128==0, N%128==0, K%64==0 (true for every launch here).
//
// 128x128x64 tile, 256 threads = 8 waves, wave sub-tile 32x64
// -> 16 x v_wmma_f32_16x16x32_bf16 per k-step per wave.
// Tiles streamed with GLOBAL_LOAD_ASYNC_TO_LDS_B128 (ASYNCcnt, double
// buffered); fragments gathered with contiguous ds_load_b128, no VALU
// conversion in the hot loop.
// =====================================================================
#define BM 128
#define BN 128
#define BK 64
#define TILE_HALF (128 * 64)            // bf16 elements per tile

__device__ __forceinline__ void async_b128(unsigned lds_off, const void* gaddr) {
    asm volatile("global_load_async_to_lds_b128 %0, %1, off"
                 :: "v"(lds_off), "v"(gaddr) : "memory");
}
__device__ __forceinline__ void wait_async0() {
    asm volatile("s_wait_asynccnt 0x0" ::: "memory");
}

// stage one 128x64 bf16 tile ([row][k], 16KB) : 4 async B128 per thread
__device__ __forceinline__ void stage_tile(const unsigned short* __restrict__ G,
                                           long ldg, int row0, int k0,
                                           unsigned lds, int tid)
{
    #pragma unroll
    for (int j = 0; j < 4; ++j) {
        int chunk = tid + j * 256;       // 1024 chunks of 8 bf16 (16B)
        int r  = chunk >> 3;
        int c8 = (chunk & 7) * 8;
        async_b128(lds + (unsigned)((r * BK + c8) * 2),
                   G + (long)(row0 + r) * ldg + (k0 + c8));
    }
}

// two contiguous 16B LDS runs -> one 16-element bf16 fragment
__device__ __forceinline__ v16bf ld_frag2(const unsigned short* p0,
                                          const unsigned short* p1)
{
    v4u lo = *(const v4u*)p0;
    v4u hi = *(const v4u*)p1;
    v8u u;
    u[0] = lo[0]; u[1] = lo[1]; u[2] = lo[2]; u[3] = lo[3];
    u[4] = hi[0]; u[5] = hi[1]; u[6] = hi[2]; u[7] = hi[3];
    return __builtin_bit_cast(v16bf, u);
}

__global__ void __launch_bounds__(256)
gemm_bf16wmma(const unsigned short* __restrict__ A, long lda,
              const unsigned short* __restrict__ B, long ldb,
              void* __restrict__ Cv, long ldc,
              int M, int N, int K,
              float alpha, int outBf16,
              const float* __restrict__ maskAdd, long ldmask,
              int lyraMask)
{
    __shared__ __align__(16) unsigned short smem[2][2][TILE_HALF]; // [buf][A/B]

    const int tid  = threadIdx.x;
    const int wave = tid >> 5;              // 0..7
    const int lane = tid & 31;
    const int wm   = (wave >> 1) * 32;      // wave row offset (0,32,64,96)
    const int wn   = (wave & 1) * 64;       // wave col offset (0,64)
    const int half = lane >> 4;             // ISA fragment lane-half
    const int l15  = lane & 15;

    const int m0 = blockIdx.y * BM;
    const int n0 = blockIdx.x * BN;

    const unsigned ldsBase = (unsigned)(unsigned long long)(uintptr_t)&smem[0][0][0];
    const unsigned ldsA0 = ldsBase;
    const unsigned ldsB0 = ldsBase + TILE_HALF * 2u;
    const unsigned bufStride = 2u * TILE_HALF * 2u;

    v8f acc[2][4];
    #pragma unroll
    for (int i = 0; i < 2; ++i)
        #pragma unroll
        for (int j = 0; j < 4; ++j)
            #pragma unroll
            for (int r = 0; r < 8; ++r) acc[i][j][r] = 0.0f;

    // prologue: stage tile 0 into buffer 0
    stage_tile(A, lda, m0, 0, ldsA0, tid);
    stage_tile(B, ldb, n0, 0, ldsB0, tid);
    wait_async0();
    __syncthreads();

    int cur = 0;
    for (int k0 = 0; k0 < K; k0 += BK) {
        const int more = (k0 + BK) < K;
        if (more) {   // stream next tile while computing this one
            stage_tile(A, lda, m0, k0 + BK, ldsA0 + (unsigned)(cur ^ 1) * bufStride, tid);
            stage_tile(B, ldb, n0, k0 + BK, ldsB0 + (unsigned)(cur ^ 1) * bufStride, tid);
        }

        const unsigned short* As = &smem[cur][0][0];
        const unsigned short* Bs = &smem[cur][1][0];

        #pragma unroll
        for (int kc = 0; kc < 2; ++kc) {            // two K=32 chunks
            v16bf afrag[2];
            #pragma unroll
            for (int im = 0; im < 2; ++im) {
                // A frag (16x32, ISA 7.12.2): lanes 0-15 K{0..7,16..23},
                // lanes 16-31 K{8..15,24..31} of this chunk
                const unsigned short* p = As + (wm + im * 16 + l15) * BK
                                             + kc * 32 + half * 8;
                afrag[im] = ld_frag2(p, p + 16);
            }
            #pragma unroll
            for (int in = 0; in < 4; ++in) {
                // B frag (32x16): lanes 0-15 K{0..7,16..23}, lanes 16-31 K{8..15,24..31}
                const unsigned short* p = Bs + (wn + in * 16 + l15) * BK
                                             + kc * 32 + half * 8;
                v16bf b = ld_frag2(p, p + 16);
                #pragma unroll
                for (int im = 0; im < 2; ++im)
                    acc[im][in] = __builtin_amdgcn_wmma_f32_16x16x32_bf16(
                        false, afrag[im], false, b, (short)0, acc[im][in],
                        false, false);
            }
        }

        if (more) wait_async0();    // next tile resident before buffer flip
        __syncthreads();
        cur ^= 1;
    }

    // ---- epilogue: C fragment layout (ISA): VGPR r -> M=r+8*half, N=l15 ----
    #pragma unroll
    for (int im = 0; im < 2; ++im)
        #pragma unroll
        for (int in = 0; in < 4; ++in)
            #pragma unroll
            for (int r = 0; r < 8; ++r) {
                int gm = m0 + wm + im * 16 + r + 8 * half;
                int gn = n0 + wn + in * 16 + l15;
                float v = acc[im][in][r] * alpha;
                if (maskAdd) v += maskAdd[(long)gm * ldmask + gn];
                if (lyraMask && gn > (kCACHE + gm)) v += 1.0f;
                long idx = (long)gm * ldc + gn;
                if (outBf16) ((unsigned short*)Cv)[idx] = (unsigned short)f2bf(v);
                else         ((float*)Cv)[idx]          = v;
            }
}

// =====================================================================
// flat fp32 -> bf16 convert (n % 4 == 0)
// =====================================================================
__global__ void __launch_bounds__(256)
cvt_bf16(const float* __restrict__ in, unsigned short* __restrict__ out, long n)
{
    long i = ((long)blockIdx.x * 256 + threadIdx.x) * 4;
    long stride = (long)gridDim.x * 256 * 4;
    for (; i < n; i += stride) {
        float4 x = *(const float4*)(in + i);
        out[i + 0] = (unsigned short)f2bf(x.x);
        out[i + 1] = (unsigned short)f2bf(x.y);
        out[i + 2] = (unsigned short)f2bf(x.z);
        out[i + 3] = (unsigned short)f2bf(x.w);
    }
}

// =====================================================================
// tiled transpose: in fp32 (R x C, row stride ldin) -> out bf16 (C x R,
// row stride ldout).  R,C multiples of 32.  Block (32,8).
// =====================================================================
__global__ void __launch_bounds__(256)
transpose_f32_to_bf16(const float* __restrict__ in, long ldin,
                      unsigned short* __restrict__ out, long ldout)
{
    __shared__ float t[32][33];
    const int r0 = blockIdx.x * 32;
    const int c0 = blockIdx.y * 32;
    const int tx = threadIdx.x, ty = threadIdx.y;
    #pragma unroll
    for (int i = ty; i < 32; i += 8)
        t[i][tx] = in[(long)(r0 + i) * ldin + c0 + tx];
    __syncthreads();
    #pragma unroll
    for (int i = ty; i < 32; i += 8)
        out[(long)(c0 + i) * ldout + r0 + tx] = (unsigned short)f2bf(t[tx][i]);
}

// =====================================================================
// Per-head RMSNorm + RoPE -> bf16.  X: (S, nh*D) fp32.
// out element at: Yb[h*headStride + s*rowStride + d]
// =====================================================================
__global__ void __launch_bounds__(128)
rmsnorm_rope_bf16(const float* __restrict__ X,
                  unsigned short* __restrict__ Yb,
                  long headStride, long rowStride,
                  const float* __restrict__ w,
                  const float* __restrict__ cosp, const float* __restrict__ sinp,
                  int nh)
{
    const int s = blockIdx.x;
    const int h = blockIdx.y;
    const int d = threadIdx.x;      // 0..127

    float xv = X[(long)s * nh * kD + (long)h * kD + d];

    __shared__ float red[kD];
    __shared__ float nn[kD];
    red[d] = xv * xv;
    __syncthreads();
    #pragma unroll
    for (int off = 64; off > 0; off >>= 1) {
        if (d < off) red[d] += red[d + off];
        __syncthreads();
    }
    float ms = red[0] * (1.0f / (float)kD);
    float n  = xv * rsqrtf(ms + kEPS) * (1.0f + w[d]);

    nn[d] = n;
    __syncthreads();
    float rot  = (d < 64) ? -nn[d + 64] : nn[d - 64];
    float outv = n * cosp[(long)s * kD + d] + rot * sinp[(long)s * kD + d];

    Yb[(long)h * headStride + (long)s * rowStride + d] = (unsigned short)f2bf(outv);
}

// =====================================================================
// Row softmax: reads fp32 scores, writes bf16 probs (and optionally the
// normalized fp32 row in place -- needed for the vanilla_w output).
// =====================================================================
__global__ void __launch_bounds__(256)
softmax_rows(float* __restrict__ W, unsigned short* __restrict__ Wb,
             int width, int writeF32)
{
    float*          row  = W  + (long)blockIdx.x * width;
    unsigned short* rowb = Wb + (long)blockIdx.x * width;
    const int tid = threadIdx.x;
    __shared__ float red[256];

    float m = -3.4e38f;
    for (int c = tid; c < width; c += 256) m = fmaxf(m, row[c]);
    red[tid] = m;
    __syncthreads();
    #pragma unroll
    for (int off = 128; off > 0; off >>= 1) {
        if (tid < off) red[tid] = fmaxf(red[tid], red[tid + off]);
        __syncthreads();
    }
    m = red[0];
    __syncthreads();

    float sum = 0.0f;
    for (int c = tid; c < width; c += 256) {
        float e = __expf(row[c] - m);
        row[c] = e;
        sum += e;
    }
    red[tid] = sum;
    __syncthreads();
    #pragma unroll
    for (int off = 128; off > 0; off >>= 1) {
        if (tid < off) red[tid] += red[tid + off];
        __syncthreads();
    }
    float inv = 1.0f / red[0];
    __syncthreads();
    for (int c = tid; c < width; c += 256) {
        float p = row[c] * inv;
        if (writeF32) row[c] = p;
        rowb[c] = (unsigned short)f2bf(p);
    }
}

// =====================================================================
// Host-side driver
// =====================================================================
static inline void launch_gemm(hipStream_t st,
                               const unsigned short* A, long lda,
                               const unsigned short* B, long ldb,
                               void* C, long ldc,
                               int M, int N, int K,
                               float alpha, int outBf16,
                               const float* maskAdd, long ldmask, int lyraMask)
{
    dim3 grid(N / BN, M / BM);
    gemm_bf16wmma<<<grid, dim3(256), 0, st>>>(A, lda, B, ldb, C, ldc, M, N, K,
                                              alpha, outBf16, maskAdd, ldmask, lyraMask);
}

static inline void launch_cvt(hipStream_t st, const float* in, unsigned short* out, long n)
{
    cvt_bf16<<<dim3(512), dim3(256), 0, st>>>(in, out, n);
}

extern "C" void kernel_launch(void* const* d_in, const int* in_sizes, int n_in,
                              void* d_out, int out_size, void* d_ws, size_t ws_size,
                              hipStream_t stream)
{
    (void)in_sizes; (void)n_in; (void)out_size; (void)ws_size;

    const float* hidden = (const float*)d_in[0];   // (S, HID)
    const float* cosp   = (const float*)d_in[1];   // (1, S, D)
    const float* sinp   = (const float*)d_in[2];   // (1, S, D)
    const float* amask  = (const float*)d_in[3];   // (1,1,S,S)
    const float* kcache = (const float*)d_in[4];   // (1, KV, CACHE, D)
    const float* vcache = (const float*)d_in[5];   // (1, KV, CACHE, D)
    const float* Wq     = (const float*)d_in[6];   // (HID, HID)
    const float* Wk     = (const float*)d_in[7];   // (KVD, HID)
    const float* Wv     = (const float*)d_in[8];   // (KVD, HID)
    const float* Wo     = (const float*)d_in[9];   // (HID, HID)
    const float* qnw    = (const float*)d_in[10];  // (D)
    const float* knw    = (const float*)d_in[11];  // (D)

    float* out  = (float*)d_out;
    float* attn = out;                             // (S, HID) fp32
    float* vw   = out + (long)kS * kHID;           // (8, S, S) fp32 vanilla_w

    // ---------------- workspace layout ----------------
    char* p = (char*)d_ws;
    auto takeH = [&](long n) { unsigned short* r = (unsigned short*)p; p += n * 2; return r; };
    auto takeF = [&](long n) { float* r = (float*)p; p += n * 4; return r; };

    unsigned short* hbf   = takeH((long)kS * kHID);        // hidden bf16
    unsigned short* wqb   = takeH((long)kHID * kHID);
    unsigned short* wkb   = takeH((long)kKVD * kHID);
    unsigned short* wvb   = takeH((long)kKVD * kHID);
    unsigned short* wob   = takeH((long)kHID * kHID);
    float*          q     = takeF((long)kS * kHID);        // fp32 projections
    float*          kproj = takeF((long)kS * kKVD);
    float*          vproj = takeF((long)kS * kKVD);
    unsigned short* qrb   = takeH((long)kS * kHID);        // roped q, bf16 (S,H,D)
    unsigned short* kcat  = takeH((long)kKV * kLW * kD);   // [cache | roped k] (KV,4096,D)
    unsigned short* lvT   = takeH((long)kKV * kD * kLW);   // [cache_v^T | v^T] (KV,D,4096)
    unsigned short* comb  = takeH((long)kS * kHID);        // attn concat, bf16
    unsigned short* pwb   = takeH((long)8 * kS * kS);      // vanilla probs bf16
    float*          lsc   = takeF((long)8 * kS * kLW);     // lyra scores fp32
    unsigned short* lwb   = takeH((long)8 * kS * kLW);     // lyra probs bf16

    // 1) one-time bf16 conversions (memory-bound pre-passes)
    launch_cvt(stream, hidden, hbf, (long)kS * kHID);
    launch_cvt(stream, Wq, wqb, (long)kHID * kHID);
    launch_cvt(stream, Wk, wkb, (long)kKVD * kHID);
    launch_cvt(stream, Wv, wvb, (long)kKVD * kHID);
    launch_cvt(stream, Wo, wob, (long)kHID * kHID);
    for (int kv = 0; kv < kKV; ++kv) {
        // key cache -> kcat rows [0, CACHE)
        launch_cvt(stream, kcache + (long)kv * kCACHE * kD,
                   kcat + (long)kv * kLW * kD, (long)kCACHE * kD);
        // value cache (CACHE x D) -> lvT cols [0, CACHE)
        transpose_f32_to_bf16<<<dim3(kCACHE / 32, kD / 32), dim3(32, 8), 0, stream>>>(
            vcache + (long)kv * kCACHE * kD, kD,
            lvT + (long)kv * kD * kLW, kLW);
    }

    // 2) projections (bf16 WMMA GEMMs, fp32 out)
    launch_gemm(stream, hbf, kHID, wqb, kHID, q,     kHID, kS, kHID, kHID, 1.f, 0, nullptr, 0, 0);
    launch_gemm(stream, hbf, kHID, wkb, kHID, kproj, kKVD, kS, kKVD, kHID, 1.f, 0, nullptr, 0, 0);
    launch_gemm(stream, hbf, kHID, wvb, kHID, vproj, kKVD, kS, kKVD, kHID, 1.f, 0, nullptr, 0, 0);

    // 3) RMSNorm + RoPE -> bf16 operands
    //    q -> qrb laid out (S, H, D): headStride=D, rowStride=HID
    rmsnorm_rope_bf16<<<dim3(kS, kH), dim3(kD), 0, stream>>>(
        q, qrb, (long)kD, (long)kHID, qnw, cosp, sinp, kH);
    //    k -> kcat rows [CACHE, CACHE+S): headStride=4096*D, rowStride=D
    rmsnorm_rope_bf16<<<dim3(kS, kKV), dim3(kD), 0, stream>>>(
        kproj, kcat + (long)kCACHE * kD, (long)kLW * kD, (long)kD, knw, cosp, sinp, kKV);
    //    v (S x D per head, strided) -> lvT cols [CACHE, CACHE+S)
    for (int kv = 0; kv < kKV; ++kv)
        transpose_f32_to_bf16<<<dim3(kS / 32, kD / 32), dim3(32, 8), 0, stream>>>(
            vproj + (long)kv * kD, kKVD,
            lvT + (long)kv * kD * kLW + kCACHE, kLW);

    // 4) vanilla attention (q heads 0..7, GQA n_rep=2)
    for (int h = 0; h < 8; ++h) {
        int kv = h >> 1;
        launch_gemm(stream,
                    qrb + (long)h * kD, kHID,                      // A: q head (S, D)
                    kcat + ((long)kv * kLW + kCACHE) * kD, kD,     // B: roped k (S, D)
                    vw + (long)h * kS * kS, kS,                    // C: scores fp32
                    kS, kS, kD, kSCALE, 0, amask, kS, 0);
    }
    softmax_rows<<<dim3(8 * kS), dim3(256), 0, stream>>>(vw, pwb, kS, 1);
    for (int h = 0; h < 8; ++h) {
        int kv = h >> 1;
        launch_gemm(stream,
                    pwb + (long)h * kS * kS, kS,                   // A: probs bf16
                    lvT + (long)kv * kD * kLW + kCACHE, kLW,       // B: v^T (D, S)
                    comb + (long)h * kD, kHID,                     // C: bf16 slice
                    kS, kD, kS, 1.f, 1, nullptr, 0, 0);
    }

    // 5) lyra attention (q heads 8..15) over [cache | k_rope] = 4096 keys
    for (int h = 0; h < 8; ++h) {
        int kv = h >> 1;
        launch_gemm(stream,
                    qrb + (long)(8 + h) * kD, kHID,
                    kcat + (long)kv * kLW * kD, kD,                // B: all 4096 keys
                    lsc + (long)h * kS * kLW, kLW,
                    kS, kLW, kD, kSCALE, 0, nullptr, 0, /*lyraMask=*/1);
    }
    softmax_rows<<<dim3(8 * kS), dim3(256), 0, stream>>>(lsc, lwb, kLW, 0);
    for (int h = 0; h < 8; ++h) {
        int kv = h >> 1;
        launch_gemm(stream,
                    lwb + (long)h * kS * kLW, kLW,                 // A: probs bf16
                    lvT + (long)kv * kD * kLW, kLW,                // B: [cache_v|v]^T
                    comb + (long)(8 + h) * kD, kHID,
                    kS, kD, kLW, 1.f, 1, nullptr, 0, 0);
    }

    // 6) output projection: attn = comb @ Wo^T  (fp32 out)
    launch_gemm(stream, comb, kHID, wob, kHID, attn, kHID, kS, kHID, kHID, 1.f, 0, nullptr, 0, 0);
}
